// Seq2Seq_85203561218332
// MI455X (gfx1250) — compile-verified, matrix operand order checked by reading, not run on hardware
//
#include <hip/hip_runtime.h>
#include <math.h>

// Problem dims (fixed by the reference)
#define B_  32
#define SE_ 256
#define SD_ 128
#define E_  1024
#define H_  1024
#define H3_ 3072
#define V_  32000

typedef __bf16 bf16_t;
typedef __attribute__((ext_vector_type(16))) __bf16 v16bf;
typedef __attribute__((ext_vector_type(8)))  __bf16 v8bf;
typedef __attribute__((ext_vector_type(8)))  float  v8f;

// ---------------------------------------------------------------------------
// WMMA helpers.  CDNA5 v_wmma_f32_16x16x32_bf16:
//   A (16x32): lane l holds row m=l&15; VGPR0..3 = K  (l>>4)*8 .. +7
//                                       VGPR4..7 = K 16+(l>>4)*8 .. +7
//   B (32x16): lane l holds col n=l&15; same K striping (row-major (N,K)
//              weight memory gives contiguous K runs -> two 16B loads).
//   C/D (16x16 f32): lane l col n=l&15, VGPR r -> row r + 8*(l>>4).
// ---------------------------------------------------------------------------
__device__ __forceinline__ v16bf load_frag(const bf16_t* p) {
  v8bf lo = *(const v8bf*)p;        // K = base .. base+7
  v8bf hi = *(const v8bf*)(p + 16); // K = base+16 .. base+23
  return __builtin_shufflevector(lo, hi, 0,1,2,3,4,5,6,7,8,9,10,11,12,13,14,15);
}

__device__ __forceinline__ v8f wmma_bf16(v16bf a, v16bf b, v8f c) {
  return __builtin_amdgcn_wmma_f32_16x16x32_bf16(false, a, false, b, (short)0, c,
                                                 false, false);
}

__device__ __forceinline__ float sigf(float x) { return 1.f / (1.f + __expf(-x)); }

// ---------------------------------------------------------------------------
// Generic GEMM: C(M,N) = A(M,K)_bf16 @ W(N,K)_bf16^T + bias
// block = 256 threads = 8 waves; wave w -> n-tile blockIdx.x*8+w;
// each wave computes MT m-tiles (B-operand reuse).  Optional (b,t) row remap
// for the final logits transpose; optional bf16 mirror of the output.
// ---------------------------------------------------------------------------
template <int MT>
__global__ __launch_bounds__(256) void gemm_bf16(
    const bf16_t* __restrict__ A, int lda,
    const bf16_t* __restrict__ W, int ldw,
    const float* __restrict__ bias,
    float* __restrict__ Cf, bf16_t* __restrict__ Cb,
    int N, int K, int remapB, int remapT) {
  const int lane  = threadIdx.x & 31;
  const int wv    = threadIdx.x >> 5;
  const int n0    = (blockIdx.x * 8 + wv) * 16;
  const int mbase = blockIdx.y * (16 * MT);
  const int lh    = lane & 15;
  const int khalf = (lane >> 4) * 8;
  const int nn    = n0 + lh;

  v8f zero = {0.f, 0.f, 0.f, 0.f, 0.f, 0.f, 0.f, 0.f};
  v8f c[MT];
#pragma unroll
  for (int i = 0; i < MT; ++i) c[i] = zero;

  const bf16_t* wrow = W + (size_t)nn * ldw + khalf;
  const bf16_t* arow[MT];
#pragma unroll
  for (int mt = 0; mt < MT; ++mt)
    arow[mt] = A + (size_t)(mbase + mt * 16 + lh) * lda + khalf;

  for (int kb = 0; kb < K; kb += 32) {
    v16bf bf = load_frag(wrow + kb);
#pragma unroll
    for (int mt = 0; mt < MT; ++mt) {
      v16bf af = load_frag(arow[mt] + kb);
      c[mt] = wmma_bf16(af, bf, c[mt]);
    }
  }

  const float bv = bias ? bias[nn] : 0.f;
  const int rowhalf = (lane >> 4) * 8;
#pragma unroll
  for (int mt = 0; mt < MT; ++mt) {
#pragma unroll
    for (int r = 0; r < 8; ++r) {
      int m = mbase + mt * 16 + rowhalf + r;
      int orow = m;
      if (remapB) {  // row m = t*remapB + b  ->  orow = b*remapT + t
        int t = m / remapB;
        int b = m - t * remapB;
        orow = b * remapT + t;
      }
      float v = c[mt][r] + bv;
      if (Cf) Cf[(size_t)orow * N + nn] = v;
      if (Cb) Cb[(size_t)orow * N + nn] = (bf16_t)v;
    }
  }
}

// ---------------------------------------------------------------------------
// Encoder GRU step. 128 blocks x 1 wave.  block = (m-tile, j-tile).
// Wave computes gh r/z/n 16x16 tiles (K=1024) for its columns, then fuses
// the gates using precomputed gi (= x@Wih^T + bih).  Masked update.
// ---------------------------------------------------------------------------
__global__ __launch_bounds__(32) void enc_step(
    const bf16_t* __restrict__ hcur_b, const float* __restrict__ hcur_f,
    const bf16_t* __restrict__ Whh, const float* __restrict__ gi_t,
    const float* __restrict__ bhh, const int* __restrict__ enc_len, int t,
    float* __restrict__ hnxt_f, bf16_t* __restrict__ hnxt_b,
    bf16_t* __restrict__ out_b) {
  const int lane = threadIdx.x & 31;
  const int mt = blockIdx.x >> 6;   // 0..1
  const int jt = blockIdx.x & 63;   // 0..63
  const int lh = lane & 15;
  const int khalf = (lane >> 4) * 8;
  const int rowhalf = (lane >> 4) * 8;
  const int m0 = mt * 16, j0 = jt * 16;

  v8f zero = {0.f, 0.f, 0.f, 0.f, 0.f, 0.f, 0.f, 0.f};
  v8f cr = zero, cz = zero, cn = zero;

  const bf16_t* arow = hcur_b + (size_t)(m0 + lh) * H_ + khalf;
  const bf16_t* wr = Whh + (size_t)(j0 + lh) * H_ + khalf;
  const bf16_t* wz = Whh + (size_t)(H_ + j0 + lh) * H_ + khalf;
  const bf16_t* wn = Whh + (size_t)(2 * H_ + j0 + lh) * H_ + khalf;

  for (int kb = 0; kb < H_; kb += 32) {
    v16bf a = load_frag(arow + kb);
    cr = wmma_bf16(a, load_frag(wr + kb), cr);
    cz = wmma_bf16(a, load_frag(wz + kb), cz);
    cn = wmma_bf16(a, load_frag(wn + kb), cn);
  }

  const int j = j0 + lh;
  const float bh_r = bhh[j], bh_z = bhh[H_ + j], bh_n = bhh[2 * H_ + j];
#pragma unroll
  for (int r = 0; r < 8; ++r) {
    int b = m0 + rowhalf + r;
    const float* g = gi_t + (size_t)b * H3_;
    float ghr = cr[r] + bh_r, ghz = cz[r] + bh_z, ghn = cn[r] + bh_n;
    float rr = sigf(g[j] + ghr);
    float zz = sigf(g[H_ + j] + ghz);
    float nv = tanhf(g[2 * H_ + j] + rr * ghn);
    float hp = hcur_f[(size_t)b * H_ + j];
    float hn = (1.f - zz) * nv + zz * hp;
    bool mk = t < enc_len[b];
    float h2 = mk ? hn : hp;
    hnxt_f[(size_t)b * H_ + j] = h2;
    hnxt_b[(size_t)b * H_ + j] = (bf16_t)h2;
    out_b[(size_t)b * H_ + j] = (bf16_t)(mk ? hn : 0.f);
  }
}

// ---------------------------------------------------------------------------
// Decoder GRU step: same tiling; wave additionally computes the ctx part of
// gi (ctx @ Wih[:,E:]^T) and adds the precomputed x part (incl. bih).
// ---------------------------------------------------------------------------
__global__ __launch_bounds__(32) void dec_step(
    const bf16_t* __restrict__ hcur_b, const float* __restrict__ hcur_f,
    const bf16_t* __restrict__ ctx_b,
    const bf16_t* __restrict__ Whh,        // (3072,1024)
    const bf16_t* __restrict__ Wic,        // dec_Wih + E, row stride 2048
    const float* __restrict__ gix_t, const float* __restrict__ bhh,
    float* __restrict__ hnxt_f, bf16_t* __restrict__ hnxt_b,
    bf16_t* __restrict__ out_b) {
  const int lane = threadIdx.x & 31;
  const int mt = blockIdx.x >> 6;
  const int jt = blockIdx.x & 63;
  const int lh = lane & 15;
  const int khalf = (lane >> 4) * 8;
  const int rowhalf = (lane >> 4) * 8;
  const int m0 = mt * 16, j0 = jt * 16;

  v8f zero = {0.f, 0.f, 0.f, 0.f, 0.f, 0.f, 0.f, 0.f};
  v8f hr = zero, hz = zero, hn_ = zero;   // gh accumulators
  v8f ir = zero, iz = zero, in_ = zero;   // gi(ctx) accumulators

  const bf16_t* hrow = hcur_b + (size_t)(m0 + lh) * H_ + khalf;
  const bf16_t* crow = ctx_b + (size_t)(m0 + lh) * H_ + khalf;
  const bf16_t* whr = Whh + (size_t)(j0 + lh) * H_ + khalf;
  const bf16_t* whz = Whh + (size_t)(H_ + j0 + lh) * H_ + khalf;
  const bf16_t* whn = Whh + (size_t)(2 * H_ + j0 + lh) * H_ + khalf;
  const bf16_t* wir = Wic + (size_t)(j0 + lh) * 2048 + khalf;
  const bf16_t* wiz = Wic + (size_t)(H_ + j0 + lh) * 2048 + khalf;
  const bf16_t* win = Wic + (size_t)(2 * H_ + j0 + lh) * 2048 + khalf;

  for (int kb = 0; kb < H_; kb += 32) {
    v16bf ah = load_frag(hrow + kb);
    v16bf ac = load_frag(crow + kb);
    hr = wmma_bf16(ah, load_frag(whr + kb), hr);
    hz = wmma_bf16(ah, load_frag(whz + kb), hz);
    hn_ = wmma_bf16(ah, load_frag(whn + kb), hn_);
    ir = wmma_bf16(ac, load_frag(wir + kb), ir);
    iz = wmma_bf16(ac, load_frag(wiz + kb), iz);
    in_ = wmma_bf16(ac, load_frag(win + kb), in_);
  }

  const int j = j0 + lh;
  const float bh_r = bhh[j], bh_z = bhh[H_ + j], bh_n = bhh[2 * H_ + j];
#pragma unroll
  for (int r = 0; r < 8; ++r) {
    int b = m0 + rowhalf + r;
    const float* g = gix_t + (size_t)b * H3_;  // x-part of gi, incl. bih
    float gir = g[j] + ir[r];
    float giz = g[H_ + j] + iz[r];
    float gin = g[2 * H_ + j] + in_[r];
    float ghr = hr[r] + bh_r, ghz = hz[r] + bh_z, ghn = hn_[r] + bh_n;
    float rr = sigf(gir + ghr);
    float zz = sigf(giz + ghz);
    float nv = tanhf(gin + rr * ghn);
    float hp = hcur_f[(size_t)b * H_ + j];
    float h2 = (1.f - zz) * nv + zz * hp;
    hnxt_f[(size_t)b * H_ + j] = h2;
    hnxt_b[(size_t)b * H_ + j] = (bf16_t)h2;
    out_b[(size_t)b * H_ + j] = (bf16_t)h2;
  }
}

// ---------------------------------------------------------------------------
// Attention helpers (tiny FLOPs, VALU is fine)
// ---------------------------------------------------------------------------
__global__ __launch_bounds__(256) void scores_kernel(
    const bf16_t* __restrict__ Kp, const float* __restrict__ q,
    float* __restrict__ sc) {
  int b = blockIdx.x, s = threadIdx.x;
  const bf16_t* kp = Kp + ((size_t)s * B_ + b) * H_;
  const float* qb = q + (size_t)b * H_;
  float acc = 0.f;
  for (int h = 0; h < H_; ++h) acc += (float)kp[h] * qb[h];
  sc[b * SE_ + s] = acc;
}

__global__ __launch_bounds__(256) void softmax256(const float* __restrict__ sc,
                                                  float* __restrict__ at) {
  int b = blockIdx.x, tid = threadIdx.x;
  __shared__ float red[256];
  float v = sc[b * SE_ + tid];
  red[tid] = v;
  __syncthreads();
  for (int off = 128; off > 0; off >>= 1) {
    if (tid < off) red[tid] = fmaxf(red[tid], red[tid + off]);
    __syncthreads();
  }
  float mx = red[0];
  __syncthreads();
  float e = __expf(v - mx);
  red[tid] = e;
  __syncthreads();
  for (int off = 128; off > 0; off >>= 1) {
    if (tid < off) red[tid] += red[tid + off];
    __syncthreads();
  }
  at[b * SE_ + tid] = e / red[0];
}

__global__ __launch_bounds__(256) void ctx_kernel(
    const float* __restrict__ at, const bf16_t* __restrict__ encO,
    bf16_t* __restrict__ ctxb) {
  int b = blockIdx.x;
  int h = blockIdx.y * 256 + threadIdx.x;
  float acc = 0.f;
  for (int s = 0; s < SE_; ++s)
    acc += at[b * SE_ + s] * (float)encO[((size_t)s * B_ + b) * H_ + h];
  ctxb[(size_t)b * H_ + h] = (bf16_t)acc;
}

// ---------------------------------------------------------------------------
// Data prep
// ---------------------------------------------------------------------------
__global__ __launch_bounds__(256) void conv_bf16(const float* __restrict__ in,
                                                 bf16_t* __restrict__ out,
                                                 size_t n) {
  size_t i = (size_t)blockIdx.x * 256 + threadIdx.x;
  size_t stride = (size_t)gridDim.x * 256;
  for (; i < n; i += stride) out[i] = (bf16_t)in[i];
}

__global__ __launch_bounds__(256) void gather_embed(
    const int* __restrict__ tok, int S, const float* __restrict__ emb,
    bf16_t* __restrict__ out) {
  int r = blockIdx.x;        // r = t*B + b  (time-major rows)
  int t = r >> 5, b = r & 31;
  int token = tok[b * S + t];
  const float* e = emb + (size_t)token * E_;
  bf16_t* o = out + (size_t)r * E_;
  for (int i = threadIdx.x; i < E_; i += 256) o[i] = (bf16_t)e[i];
}

// ---------------------------------------------------------------------------
extern "C" void kernel_launch(void* const* d_in, const int* in_sizes, int n_in,
                              void* d_out, int out_size, void* d_ws,
                              size_t ws_size, hipStream_t stream) {
  const int* enc_in = (const int*)d_in[0];
  const int* dec_in = (const int*)d_in[1];
  const int* enc_len = (const int*)d_in[2];
  // d_in[3] decoder_lengths: unused by the reference
  const float* emb = (const float*)d_in[4];
  const float* eWih = (const float*)d_in[5];
  const float* eWhh = (const float*)d_in[6];
  const float* ebih = (const float*)d_in[7];
  const float* ebhh = (const float*)d_in[8];
  const float* dWih = (const float*)d_in[9];
  const float* dWhh = (const float*)d_in[10];
  const float* dbih = (const float*)d_in[11];
  const float* dbhh = (const float*)d_in[12];
  const float* kW = (const float*)d_in[13];
  const float* kb = (const float*)d_in[14];
  const float* qW = (const float*)d_in[15];
  const float* qb = (const float*)d_in[16];
  const float* lW = (const float*)d_in[17];
  const float* lb = (const float*)d_in[18];
  float* out = (float*)d_out;

  char* ws = (char*)d_ws;
  size_t off = 0;
  auto alloc = [&](size_t bytes) -> char* {
    char* p = ws + off;
    off += (bytes + 255) & ~(size_t)255;
    return p;
  };

  bf16_t* eWih_b = (bf16_t*)alloc((size_t)H3_ * E_ * 2);
  bf16_t* eWhh_b = (bf16_t*)alloc((size_t)H3_ * H_ * 2);
  bf16_t* dWih_b = (bf16_t*)alloc((size_t)H3_ * (E_ + H_) * 2);
  bf16_t* dWhh_b = (bf16_t*)alloc((size_t)H3_ * H_ * 2);
  bf16_t* kW_b = (bf16_t*)alloc((size_t)H_ * H_ * 2);
  bf16_t* qW_b = (bf16_t*)alloc((size_t)H_ * H_ * 2);
  bf16_t* lW_b = (bf16_t*)alloc((size_t)V_ * H_ * 2);
  bf16_t* enc_x_b = (bf16_t*)alloc((size_t)SE_ * B_ * E_ * 2);
  bf16_t* dec_x_b = (bf16_t*)alloc((size_t)SD_ * B_ * E_ * 2);
  float* enc_gi = (float*)alloc((size_t)SE_ * B_ * H3_ * 4);
  float* dec_gix = (float*)alloc((size_t)SD_ * B_ * H3_ * 4);
  bf16_t* enc_out_b = (bf16_t*)alloc((size_t)SE_ * B_ * H_ * 2);
  bf16_t* Kproj_b = (bf16_t*)alloc((size_t)SE_ * B_ * H_ * 2);
  float* hA_f = (float*)alloc((size_t)B_ * H_ * 4);
  float* hB_f = (float*)alloc((size_t)B_ * H_ * 4);
  bf16_t* hA_b = (bf16_t*)alloc((size_t)B_ * H_ * 2);
  bf16_t* hB_b = (bf16_t*)alloc((size_t)B_ * H_ * 2);
  float* qbuf = (float*)alloc((size_t)B_ * H_ * 4);
  float* sc = (float*)alloc((size_t)B_ * SE_ * 4);
  float* at = (float*)alloc((size_t)B_ * SE_ * 4);
  bf16_t* ctx_b = (bf16_t*)alloc((size_t)B_ * H_ * 2);
  bf16_t* dec_out_b = (bf16_t*)alloc((size_t)SD_ * B_ * H_ * 2);
  (void)ws_size; (void)in_sizes; (void)n_in; (void)out_size;

  // Weight conversions to bf16 (re-done every call; ~13us of HBM traffic)
  conv_bf16<<<2048, 256, 0, stream>>>(eWih, eWih_b, (size_t)H3_ * E_);
  conv_bf16<<<2048, 256, 0, stream>>>(eWhh, eWhh_b, (size_t)H3_ * H_);
  conv_bf16<<<2048, 256, 0, stream>>>(dWih, dWih_b, (size_t)H3_ * (E_ + H_));
  conv_bf16<<<2048, 256, 0, stream>>>(dWhh, dWhh_b, (size_t)H3_ * H_);
  conv_bf16<<<1024, 256, 0, stream>>>(kW, kW_b, (size_t)H_ * H_);
  conv_bf16<<<1024, 256, 0, stream>>>(qW, qW_b, (size_t)H_ * H_);
  conv_bf16<<<4096, 256, 0, stream>>>(lW, lW_b, (size_t)V_ * H_);

  // Embedding gathers (time-major rows t*B+b)
  gather_embed<<<SE_ * B_, 256, 0, stream>>>(enc_in, SE_, emb, enc_x_b);
  gather_embed<<<SD_ * B_, 256, 0, stream>>>(dec_in, SD_, emb, dec_x_b);

  // Hoisted input-to-hidden projections (big WMMA GEMMs)
  gemm_bf16<4><<<dim3(H3_ / 128, (SE_ * B_) / 64), 256, 0, stream>>>(
      enc_x_b, E_, eWih_b, E_, ebih, enc_gi, nullptr, H3_, E_, 0, 0);
  gemm_bf16<4><<<dim3(H3_ / 128, (SD_ * B_) / 64), 256, 0, stream>>>(
      dec_x_b, E_, dWih_b, E_ + H_, dbih, dec_gix, nullptr, H3_, E_, 0, 0);

  // h0 = 0
  hipMemsetAsync(hA_f, 0, (size_t)B_ * H_ * 4, stream);
  hipMemsetAsync(hA_b, 0, (size_t)B_ * H_ * 2, stream);

  // Encoder recurrence
  float* hc_f = hA_f; bf16_t* hc_b = hA_b;
  float* hn_f = hB_f; bf16_t* hn_b = hB_b;
  for (int t = 0; t < SE_; ++t) {
    enc_step<<<128, 32, 0, stream>>>(
        hc_b, hc_f, eWhh_b, enc_gi + (size_t)t * B_ * H3_, ebhh, enc_len, t,
        hn_f, hn_b, enc_out_b + (size_t)t * B_ * H_);
    float* tf = hc_f; hc_f = hn_f; hn_f = tf;
    bf16_t* tb = hc_b; hc_b = hn_b; hn_b = tb;
  }

  // Attention keys: Kproj = enc_out @ kW^T + kb (bf16, zero rows -> kb rows,
  // matching the reference's unmasked softmax semantics)
  gemm_bf16<4><<<dim3(H_ / 128, (SE_ * B_) / 64), 256, 0, stream>>>(
      enc_out_b, H_, kW_b, H_, kb, nullptr, Kproj_b, H_, H_, 0, 0);

  // Decoder recurrence with attention
  for (int t = 0; t < SD_; ++t) {
    gemm_bf16<2><<<dim3(H_ / 128, 1), 256, 0, stream>>>(
        hc_b, H_, qW_b, H_, qb, qbuf, nullptr, H_, H_, 0, 0);
    scores_kernel<<<B_, SE_, 0, stream>>>(Kproj_b, qbuf, sc);
    softmax256<<<B_, SE_, 0, stream>>>(sc, at);
    ctx_kernel<<<dim3(B_, H_ / 256), 256, 0, stream>>>(at, enc_out_b, ctx_b);
    dec_step<<<128, 32, 0, stream>>>(
        hc_b, hc_f, ctx_b, dWhh_b, dWih_b + E_,
        dec_gix + (size_t)t * B_ * H3_, dbhh, hn_f, hn_b,
        dec_out_b + (size_t)t * B_ * H_);
    float* tf = hc_f; hc_f = hn_f; hn_f = tf;
    bf16_t* tb = hc_b; hc_b = hn_b; hn_b = tb;
  }

  // Logits: (SD*B,H) @ (V,H)^T + b, written transposed to (B,SD,V)
  gemm_bf16<4><<<dim3(V_ / 128, (SD_ * B_) / 64), 256, 0, stream>>>(
      dec_out_b, H_, lW_b, H_, lb, out, nullptr, V_, H_, B_, SD_);
}
